// SpatialAttentionBlock_64914135711912
// MI455X (gfx1250) — matrix operand order repaired and checked by reading, not compiled, not run on hardware
//
#include <hip/hip_runtime.h>
#include <hip/hip_bf16.h>

// SpatialAttentionBlock for MI455X (gfx1250, wave32, WMMA).
// Pipeline:
//   k0a: zero q/k fragment arrays (K-dim padding for 16x16x32 WMMA)
//   k0b: pre-pack 1x1-conv weights into WMMA B-fragment order (bf16)
//   k1 : conv q (1x3) + conv k (3x1), raw fp32
//   k2 : FRN (rms over spatial) + Mish, pack q/k into WMMA fragment order
//   k3 : conv v (1x1) as bf16 WMMA GEMM, outputs directly in V-fragment order
//   k4a: softmax row stats (rowmax, 1/rowsum) via QK^T WMMA, shuffle-free loop
//   k4b: PV accumulation: P=exp(S-max) split across waves, shared via LDS,
//        O=P@V (WMMA), coalesced epilogue through LDS transpose

typedef __attribute__((ext_vector_type(16))) __bf16 v16bf;
typedef __attribute__((ext_vector_type(8)))  __bf16 v8bf;
typedef __attribute__((ext_vector_type(8)))  float  v8f;

#define Bb   4
#define Cc   256
#define C16c 16
#define Hh   64
#define Wd   64
#define Nn   4096   // H*W

// ---- workspace layout (bytes) ----
#define QFRAG_OFF  ((size_t)0)          // 1 MB  [b][jtile][lane][half] bf16 (A-layout)
#define KFRAG_OFF  ((size_t)0x100000)   // 1 MB  [b][ntile][lane][half] bf16 (B-layout)
#define VFRAG_OFF  ((size_t)0x200000)   // 8 MB  [b][npair][ctile][lane][half] bf16 (B-layout)
#define WVFRAG_OFF ((size_t)0xA00000)   // 128 KB [ctile][kstep][lane][half] bf16 (B-layout)
#define QRAW_OFF   ((size_t)0xA40000)   // 1 MB fp32
#define KRAW_OFF   ((size_t)0xB40000)   // 1 MB fp32
#define SMAX_OFF   ((size_t)0xC40000)   // 64 KB fp32 [b][j] row max
#define SINV_OFF   ((size_t)0xC50000)   // 64 KB fp32 [b][j] 1/row sum

// ---------------- k0a: zero fill ----------------
__global__ void zero_u32_kernel(unsigned int* __restrict__ p, int count) {
    int i = blockIdx.x * 256 + threadIdx.x;
    if (i < count) p[i] = 0u;
}

// ---------------- k0b: pack wv into B-fragment order ----------------
// B[k][col] = wv[c0+col][ci0+k]; lane = col + 16*h, half<8: k = 8h+half, else k = 16+8h+(half-8)
__global__ void build_wvfrag_kernel(const float* __restrict__ wv,
                                    __bf16* __restrict__ wvfrag) {
    int tid  = blockIdx.x * 256 + threadIdx.x;   // 0..65535
    int half = tid & 15;
    int lane = (tid >> 4) & 31;
    int s    = (tid >> 9) & 7;
    int ct   = tid >> 12;
    int c    = ct * 16 + (lane & 15);
    int ci   = s * 32 + (half < 8 ? (lane >> 4) * 8 + half : 16 + (lane >> 4) * 8 + (half - 8));
    wvfrag[tid] = (__bf16)wv[c * Cc + ci];
}

// ---------------- k1: conv q (1x3, pad W) + conv k (3x1, pad H) ----------------
__global__ void conv_qk_kernel(const float* __restrict__ x,
                               const float* __restrict__ wq, const float* __restrict__ bq,
                               const float* __restrict__ wk, const float* __restrict__ bk,
                               float* __restrict__ qraw, float* __restrict__ kraw) {
    int n = blockIdx.x * 256 + threadIdx.x;   // pixel
    int b = blockIdx.y;
    int w = n & (Wd - 1), h = n >> 6;
    float qa[C16c], ka[C16c];
#pragma unroll
    for (int co = 0; co < C16c; ++co) { qa[co] = bq[co]; ka[co] = bk[co]; }
    const float* xb = x + (size_t)b * Cc * Nn + n;
    for (int ci = 0; ci < Cc; ++ci) {
        const float* xc = xb + (size_t)ci * Nn;
        float xm = xc[0];
        float xl = (w > 0)      ? xc[-1]   : 0.f;
        float xr = (w < Wd - 1) ? xc[1]    : 0.f;
        float xu = (h > 0)      ? xc[-Wd]  : 0.f;
        float xd = (h < Hh - 1) ? xc[Wd]   : 0.f;
        const float* wqc = wq + ci * 3;   // + co*Cc*3
        const float* wkc = wk + ci * 3;
#pragma unroll
        for (int co = 0; co < C16c; ++co) {
            qa[co] += wqc[co * Cc * 3 + 0] * xl + wqc[co * Cc * 3 + 1] * xm
                    + wqc[co * Cc * 3 + 2] * xr;
            ka[co] += wkc[co * Cc * 3 + 0] * xu + wkc[co * Cc * 3 + 1] * xm
                    + wkc[co * Cc * 3 + 2] * xd;
        }
    }
#pragma unroll
    for (int co = 0; co < C16c; ++co) {
        qraw[((size_t)(b * C16c + co)) * Nn + n] = qa[co];
        kraw[((size_t)(b * C16c + co)) * Nn + n] = ka[co];
    }
}

// ---------------- k2: FRN + Mish + pack into fragment order ----------------
__global__ void frn_mish_pack_kernel(const float* __restrict__ qraw,
                                     const float* __restrict__ kraw,
                                     const float* __restrict__ eps_q,
                                     const float* __restrict__ eps_k,
                                     __bf16* __restrict__ qfrag,
                                     __bf16* __restrict__ kfrag) {
    int id     = blockIdx.x;        // 0..127
    int branch = id >> 6;           // 0 = q, 1 = k
    int b      = (id >> 4) & 3;
    int co     = id & 15;
    const float* raw  = (branch ? kraw : qraw) + ((size_t)(b * C16c + co)) * Nn;
    const float* epsp = branch ? eps_k : eps_q;
    __bf16*      frag = branch ? kfrag : qfrag;

    __shared__ float red[256];
    int t = threadIdx.x;
    float ss = 0.f;
    for (int i = t; i < Nn; i += 256) { float v = raw[i]; ss += v * v; }
    red[t] = ss;
    __syncthreads();
    for (int s = 128; s > 0; s >>= 1) {
        if (t < s) red[t] += red[t + s];
        __syncthreads();
    }
    float scale = rsqrtf(red[0] * (1.0f / (float)Nn) + fabsf(epsp[co]));

    for (int i = t; i < Nn; i += 256) {
        float y  = raw[i] * scale;
        float sp = (y > 20.f) ? y : log1pf(__expf(y));
        float mv = y * tanhf(sp);
        int tile = i >> 4, lm = i & 15;
        int lane, half;
        if (branch == 0) {  // A-layout: lanes<16 -> ch 0..7, lanes>=16 -> ch 8..15
            lane = (co < 8) ? lm : (16 + lm);
            half = co & 7;
        } else {            // B-layout: lanes<16 -> ch 0..7, lanes>=16 -> ch 8..15
            lane = (co < 8) ? lm : (16 + lm);
            half = co & 7;
        }
        frag[((size_t)(b * 256 + tile) * 32 + lane) * 16 + half] = (__bf16)mv;
    }
}

// ---------------- k3: conv v (1x1) as WMMA GEMM ----------------
__global__ __launch_bounds__(256) void conv_v_wmma_kernel(
        const float* __restrict__ x, const __bf16* __restrict__ wvfrag,
        const float* __restrict__ bv, __bf16* __restrict__ vfrag) {
    int blk  = blockIdx.x;          // b*256 + tile
    int b    = blk >> 8;
    int tile = blk & 255;           // 16-pixel tile
    int n0   = tile * 16;
    int t    = threadIdx.x;

    __shared__ alignas(32) __bf16 xs[16 * 264];   // [pixel m][ci], padded stride

    const float* xb = x + (size_t)b * Cc * Nn + n0;
    {
        int m  = t & 15;
        int c0 = t >> 4;
        for (int i = 0; i < 16; ++i) {
            int ci = c0 + i * 16;
            xs[m * 264 + ci] = (__bf16)xb[(size_t)ci * Nn + m];
        }
    }
    __syncthreads();

    int wave = t >> 5, l = t & 31;
    int m = l & 15, kbase = (l < 16) ? 0 : 8;
    int ct0 = wave * 2, ct1 = wave * 2 + 1;
    v8f acc0 = {}, acc1 = {};

#pragma unroll
    for (int s = 0; s < 8; ++s) {
        v8bf lo = *(const v8bf*)&xs[m * 264 + s * 32 + kbase];
        v8bf hi = *(const v8bf*)&xs[m * 264 + s * 32 + 16 + kbase];
        v16bf a;
#pragma unroll
        for (int i = 0; i < 8; ++i) { a[i] = lo[i]; a[8 + i] = hi[i]; }
        v16bf b0 = *(const v16bf*)&wvfrag[(((size_t)(ct0 * 8 + s)) * 32 + l) * 16];
        v16bf b1 = *(const v16bf*)&wvfrag[(((size_t)(ct1 * 8 + s)) * 32 + l) * 16];
        acc0 = __builtin_amdgcn_wmma_f32_16x16x32_bf16(false, a, false, b0,
                                                       (short)0, acc0, false, false);
        acc1 = __builtin_amdgcn_wmma_f32_16x16x32_bf16(false, a, false, b1,
                                                       (short)0, acc1, false, false);
    }

    float bvv0 = bv[ct0 * 16 + (l & 15)];
    float bvv1 = bv[ct1 * 16 + (l & 15)];
    // D element (M = g + 8*(l>>4) = local pixel, N = l&15 = local channel).
    // V B-fragment: k = (tile&1)*16 + M, lane' = (l&15)+16*(tile&1), half' = M.
    int npair = tile >> 1;
    int lanep = (l & 15) + 16 * (tile & 1);
    size_t base0 = (((size_t)(b * 128 + npair) * 16 + ct0) * 32 + lanep) * 16 + (l >> 4) * 8;
    size_t base1 = (((size_t)(b * 128 + npair) * 16 + ct1) * 32 + lanep) * 16 + (l >> 4) * 8;
    v8bf o0, o1;
#pragma unroll
    for (int g = 0; g < 8; ++g) {
        o0[g] = (__bf16)(acc0[g] + bvv0);
        o1[g] = (__bf16)(acc1[g] + bvv1);
    }
    *(v8bf*)&vfrag[base0] = o0;
    *(v8bf*)&vfrag[base1] = o1;
}

// ---------------- k4a: softmax row statistics ----------------
// One wave per 16-query tile: pass 1 row max, pass 2 row sum of exp(S-max).
// No shuffles inside the loops; 8 cross-lane reductions per pass at the end.
__global__ __launch_bounds__(256) void softmax_stats_kernel(
        const __bf16* __restrict__ qfrag, const __bf16* __restrict__ kfrag,
        float* __restrict__ smax, float* __restrict__ sinv) {
    int blk = blockIdx.x;           // b*32 + jtg
    int b = blk >> 5, jtg = blk & 31;
    int t = threadIdx.x, wave = t >> 5, l = t & 31;
    int jt = jtg * 8 + wave;

    v16bf qf = *(const v16bf*)&qfrag[((size_t)(b * 256 + jt) * 32 + l) * 16];

    v8f mx;
#pragma unroll
    for (int g = 0; g < 8; ++g) mx[g] = -1e30f;

    for (int nt = 0; nt < 256; ++nt) {
        v16bf kf = *(const v16bf*)&kfrag[((size_t)(b * 256 + nt) * 32 + l) * 16];
        v8f z = {};
        v8f s = __builtin_amdgcn_wmma_f32_16x16x32_bf16(false, qf, false, kf,
                                                        (short)0, z, false, false);
#pragma unroll
        for (int g = 0; g < 8; ++g) mx[g] = fmaxf(mx[g], s[g]);
    }
#pragma unroll
    for (int g = 0; g < 8; ++g) {
        float v = mx[g];
        v = fmaxf(v, __shfl_xor(v, 1, 32));
        v = fmaxf(v, __shfl_xor(v, 2, 32));
        v = fmaxf(v, __shfl_xor(v, 4, 32));
        v = fmaxf(v, __shfl_xor(v, 8, 32));
        mx[g] = v;
    }

    v8f sum = {};
    for (int nt = 0; nt < 256; ++nt) {
        v16bf kf = *(const v16bf*)&kfrag[((size_t)(b * 256 + nt) * 32 + l) * 16];
        v8f z = {};
        v8f s = __builtin_amdgcn_wmma_f32_16x16x32_bf16(false, qf, false, kf,
                                                        (short)0, z, false, false);
#pragma unroll
        for (int g = 0; g < 8; ++g) sum[g] += __expf(s[g] - mx[g]);
    }
#pragma unroll
    for (int g = 0; g < 8; ++g) {
        float v = sum[g];
        v += __shfl_xor(v, 1, 32);
        v += __shfl_xor(v, 2, 32);
        v += __shfl_xor(v, 4, 32);
        v += __shfl_xor(v, 8, 32);
        sum[g] = v;
    }

    if ((l & 15) == 0) {
#pragma unroll
        for (int g = 0; g < 8; ++g) {
            int j = jt * 16 + g + 8 * (l >> 4);
            smax[(size_t)b * Nn + j] = mx[g];
            sinv[(size_t)b * Nn + j] = 1.f / sum[g];
        }
    }
}

// ---------------- k4b: PV accumulation + epilogue ----------------
// Block = 16 queries x 256 channels; wave w owns channels [32w, 32w+32).
// All waves compute S (cheap WMMA); exp work is split: wave w exponentiates
// rows {w, w+8} into a double-buffered shared P tile; one barrier/iteration.
__global__ __launch_bounds__(256) void attention_pv_kernel(
        const __bf16* __restrict__ qfrag, const __bf16* __restrict__ kfrag,
        const __bf16* __restrict__ vfrag,
        const float* __restrict__ smax, const float* __restrict__ sinv,
        const float* __restrict__ x, const float* __restrict__ gamma,
        float* __restrict__ out) {
    int blk = blockIdx.x;           // b*256 + jt
    int b = blk >> 8, jt = blk & 255;
    int t = threadIdx.x, wave = t >> 5, l = t & 31;
    int m = l & 15, kbase = (l < 16) ? 0 : 16;

    __shared__ alignas(32) __bf16 psbuf[2][16 * 32];   // shared P tile (double buffer)
    __shared__ alignas(16) float  otile[8][32 * 17];   // per-wave transpose buffer

    v16bf qf = *(const v16bf*)&qfrag[((size_t)(b * 256 + jt) * 32 + l) * 16];
    // row stats for this wave's assigned P rows: row = wave + 8*(l>>4)
    float mstat = smax[(size_t)b * Nn + jt * 16 + wave + 8 * (l >> 4)];

    int ctg0 = wave * 2, ctg1 = wave * 2 + 1;
    v8f acc0 = {}, acc1 = {};

    for (int np = 0; np < 128; ++np) {
        int t0 = 2 * np, t1 = 2 * np + 1;
        v16bf kf0 = *(const v16bf*)&kfrag[((size_t)(b * 256 + t0) * 32 + l) * 16];
        v16bf kf1 = *(const v16bf*)&kfrag[((size_t)(b * 256 + t1) * 32 + l) * 16];
        if (np < 127) {
            __builtin_prefetch(
                &vfrag[(((size_t)(b * 128 + np + 1) * 16 + ctg0) * 32 + l) * 16], 0, 1);
        }
        v8f z = {};
        v8f s0 = __builtin_amdgcn_wmma_f32_16x16x32_bf16(false, qf, false, kf0,
                                                         (short)0, z, false, false);
        v8f s1 = __builtin_amdgcn_wmma_f32_16x16x32_bf16(false, qf, false, kf1,
                                                         (short)0, z, false, false);
        // uniform (per-wave) select of this wave's row from the D-layout
        float sv0 = s0[0], sv1 = s1[0];
#pragma unroll
        for (int g = 1; g < 8; ++g) {
            sv0 = (wave == g) ? s0[g] : sv0;
            sv1 = (wave == g) ? s1[g] : sv1;
        }
        __bf16* ps = &psbuf[np & 1][0];
        int row = wave + 8 * (l >> 4);
        ps[row * 32 + (l & 15)]      = (__bf16)__expf(sv0 - mstat);
        ps[row * 32 + 16 + (l & 15)] = (__bf16)__expf(sv1 - mstat);
        __syncthreads();
        v8bf plo = *(const v8bf*)&ps[m * 32 + kbase];
        v8bf phi = *(const v8bf*)&ps[m * 32 + 8 + kbase];
        v16bf pf;
#pragma unroll
        for (int i = 0; i < 8; ++i) { pf[i] = plo[i]; pf[8 + i] = phi[i]; }
        v16bf vf0 = *(const v16bf*)&vfrag[(((size_t)(b * 128 + np) * 16 + ctg0) * 32 + l) * 16];
        v16bf vf1 = *(const v16bf*)&vfrag[(((size_t)(b * 128 + np) * 16 + ctg1) * 32 + l) * 16];
        acc0 = __builtin_amdgcn_wmma_f32_16x16x32_bf16(false, pf, false, vf0,
                                                       (short)0, acc0, false, false);
        acc1 = __builtin_amdgcn_wmma_f32_16x16x32_bf16(false, pf, false, vf1,
                                                       (short)0, acc1, false, false);
    }

    // epilogue: normalize + gamma, transpose through LDS, coalesced +x and store
    float gm = gamma[0];
    float* ot = &otile[wave][0];
#pragma unroll
    for (int g = 0; g < 8; ++g) {
        int jl = g + 8 * (l >> 4);
        float inv = sinv[(size_t)b * Nn + jt * 16 + jl];
        ot[(l & 15) * 17 + jl]        = gm * acc0[g] * inv;
        ot[(16 + (l & 15)) * 17 + jl] = gm * acc1[g] * inv;
    }
    asm volatile("s_wait_dscnt 0" ::: "memory");   // wave-private LDS transpose
    int c = wave * 32 + l;
    const float* xrow = x + ((size_t)b * Cc + c) * Nn + jt * 16;
    float*       orow = out + ((size_t)b * Cc + c) * Nn + jt * 16;
#pragma unroll
    for (int i = 0; i < 16; i += 4) {
        float4 xv = *(const float4*)&xrow[i];
        float4 ov;
        ov.x = ot[l * 17 + i + 0] + xv.x;
        ov.y = ot[l * 17 + i + 1] + xv.y;
        ov.z = ot[l * 17 + i + 2] + xv.z;
        ov.w = ot[l * 17 + i + 3] + xv.w;
        *(float4*)&orow[i] = ov;
    }
}

// ---------------- launcher ----------------
extern "C" void kernel_launch(void* const* d_in, const int* in_sizes, int n_in,
                              void* d_out, int out_size, void* d_ws, size_t ws_size,
                              hipStream_t stream) {
    const float* x     = (const float*)d_in[0];
    const float* wq    = (const float*)d_in[1];
    const float* bq    = (const float*)d_in[2];
    const float* wk    = (const float*)d_in[3];
    const float* bk    = (const float*)d_in[4];
    const float* wv    = (const float*)d_in[5];
    const float* bv    = (const float*)d_in[6];
    const float* gamma = (const float*)d_in[7];
    const float* eps_q = (const float*)d_in[8];
    const float* eps_k = (const float*)d_in[9];
    float* out = (float*)d_out;

    char* ws = (char*)d_ws;
    __bf16* qfrag  = (__bf16*)(ws + QFRAG_OFF);
    __bf16* kfrag  = (__bf16*)(ws + KFRAG_OFF);
    __bf16* vfrag  = (__bf16*)(ws + VFRAG_OFF);
    __bf16* wvfrag = (__bf16*)(ws + WVFRAG_OFF);
    float*  qraw   = (float*)(ws + QRAW_OFF);
    float*  kraw   = (float*)(ws + KRAW_OFF);
    float*  smax   = (float*)(ws + SMAX_OFF);
    float*  sinv   = (float*)(ws + SINV_OFF);

    // k0a: zero q/k fragment arrays (2 MB = 524288 u32)
    zero_u32_kernel<<<2048, 256, 0, stream>>>((unsigned int*)(ws + QFRAG_OFF), 524288);
    // k0b: weight fragments (65536 elems)
    build_wvfrag_kernel<<<256, 256, 0, stream>>>(wv, wvfrag);
    // k1: q/k convs
    conv_qk_kernel<<<dim3(Nn / 256, Bb), 256, 0, stream>>>(x, wq, bq, wk, bk, qraw, kraw);
    // k2: FRN + Mish + pack
    frn_mish_pack_kernel<<<128, 256, 0, stream>>>(qraw, kraw, eps_q, eps_k, qfrag, kfrag);
    // k3: v conv GEMM (WMMA)
    conv_v_wmma_kernel<<<Bb * 256, 256, 0, stream>>>(x, wvfrag, bv, vfrag);
    // k4a: softmax row stats (WMMA)
    softmax_stats_kernel<<<Bb * 32, 256, 0, stream>>>(qfrag, kfrag, smax, sinv);
    // k4b: PV + epilogue (WMMA)
    attention_pv_kernel<<<Bb * 256, 256, 0, stream>>>(qfrag, kfrag, vfrag, smax, sinv,
                                                      x, gamma, out);
}
